// Gate_23957327577370
// MI455X (gfx1250) — 8-bit matrix operand order repaired and checked by reading, compiled, not run on hardware
//
#include <hip/hip_runtime.h>
#include <stdint.h>


typedef __attribute__((ext_vector_type(8))) int   v8i;
typedef __attribute__((ext_vector_type(4))) float f4;

#define C_DIM 512
#define B_DIM 16
#define HW    16384

__device__ __forceinline__ int sgn8(float x) {
    return x > 0.f ? 1 : (x < 0.f ? -1 : 0);
}
__device__ __forceinline__ int pack4(int b0, int b1, int b2, int b3) {
    return (b0 & 0xFF) | ((b1 & 0xFF) << 8) | ((b2 & 0xFF) << 16) | ((b3 & 0xFF) << 24);
}

__global__ void pool_kernel(const float* __restrict__ x, float* __restrict__ p) {
    __shared__ float sdata[256];
    const int row = blockIdx.x;
    const f4* x4 = (const f4*)(x + (size_t)row * HW);
    const int t = threadIdx.x;
    float s = 0.f;
#pragma unroll
    for (int i = 0; i < 16; ++i) {
        f4 v = __builtin_nontemporal_load(x4 + t + i * 256);
        s += (v.x + v.y) + (v.z + v.w);
    }
    sdata[t] = s;
    __syncthreads();
    for (int off = 128; off > 0; off >>= 1) {
        if (t < off) sdata[t] += sdata[t + off];
        __syncthreads();
    }
    if (t == 0) p[row] = sdata[0] * (1.0f / HW);
}

__global__ void pack_kernel(const float* __restrict__ w1, const float* __restrict__ w2,
                            int* __restrict__ sw1, int* __restrict__ sw2,
                            float* __restrict__ scale1, float* __restrict__ scale2) {
    __shared__ float sdata[128];
    const int  j      = blockIdx.x & (C_DIM - 1);
    const bool second = blockIdx.x >= C_DIM;
    const float* w    = second ? w2 : w1;
    int*   sw         = second ? sw2 : sw1;
    float* scale      = second ? scale2 : scale1;
    const int t = threadIdx.x;

    const f4 v = ((const f4*)(w + j * C_DIM))[t];
    sw[(j * C_DIM >> 2) + t] = pack4(sgn8(v.x), sgn8(v.y), sgn8(v.z), sgn8(v.w));
    sdata[t] = fabsf(v.x) + fabsf(v.y) + fabsf(v.z) + fabsf(v.w);
    __syncthreads();
    for (int off = 64; off > 0; off >>= 1) {
        if (t < off) sdata[t] += sdata[t + off];
        __syncthreads();
    }
    if (t == 0) scale[j] = sdata[0] * (1.0f / C_DIM);
}

__global__ void layer1_kernel(const float* __restrict__ p,
                              const float* __restrict__ move1,
                              const int*   __restrict__ sw1,
                              const float* __restrict__ scale1,
                              const float* __restrict__ gamma,
                              const float* __restrict__ bn_beta,
                              const float* __restrict__ lbias,
                              const float* __restrict__ prelu_a,
                              const float* __restrict__ move2,
                              signed char* __restrict__ xb2) {
    const int lane = threadIdx.x;
    const int half = lane >> 4;
    const int mn   = lane & 15;
    const int j    = (blockIdx.x << 4) + mn;

    v8i acc = {0, 0, 0, 0, 0, 0, 0, 0};
#pragma unroll
    for (int kb = 0; kb < C_DIM; kb += 64) {
        v8i a, b;
#pragma unroll
        for (int v = 0; v < 8; ++v) {
            const int ka = kb + ((v >> 2) << 5) + (half << 4) + ((v & 3) << 2);
            const float* pp = p + mn * C_DIM + ka;
            const float* mb = move1 + ka;
            a[v] = pack4(sgn8(pp[0] + mb[0]), sgn8(pp[1] + mb[1]),
                         sgn8(pp[2] + mb[2]), sgn8(pp[3] + mb[3]));
            const int kv = kb + ((v >> 2) << 5) + (half << 4) + ((v & 3) << 2);
            b[v] = sw1[(j * C_DIM + kv) >> 2];
        }
        acc = __builtin_amdgcn_wmma_i32_16x16x64_iu8(true, a, true, b, acc, false, false);
    }

    const float sc = scale1[j];
    float h[8], s = 0.f, s2 = 0.f;
#pragma unroll
    for (int v = 0; v < 8; ++v) {
        h[v] = (float)acc[v] * sc;
        s  += h[v];
        s2 += h[v] * h[v];
    }
    s  += __shfl_xor(s, 16);
    s2 += __shfl_xor(s2, 16);
    const float mu   = s * (1.0f / B_DIM);
    const float var  = s2 * (1.0f / B_DIM) - mu * mu;
    const float rstd = rsqrtf(var + 1e-5f);
    const float g  = gamma[j];
    const float bb = bn_beta[j] + lbias[j];
    const float aj = prelu_a[j];
    const float m2 = move2[j];
#pragma unroll
    for (int v = 0; v < 8; ++v) {
        float hv = (h[v] - mu) * rstd * g + bb;
        hv = hv >= 0.f ? hv : aj * hv;
        xb2[(v + (half << 3)) * C_DIM + j] = (signed char)sgn8(hv + m2);
    }
}

__global__ void layer2_kernel(const signed char* __restrict__ xb2,
                              const int*   __restrict__ sw2,
                              const float* __restrict__ scale2,
                              float* __restrict__ out) {
    const int lane = threadIdx.x;
    const int half = lane >> 4;
    const int mn   = lane & 15;
    const int c    = (blockIdx.x << 4) + mn;
    const int* xa  = (const int*)xb2;

    v8i acc = {0, 0, 0, 0, 0, 0, 0, 0};
#pragma unroll
    for (int kb = 0; kb < C_DIM; kb += 64) {
        v8i a, b;
#pragma unroll
        for (int v = 0; v < 8; ++v) {
            const int ka = kb + ((v >> 2) << 5) + (half << 4) + ((v & 3) << 2);
            a[v] = xa[(mn * C_DIM + ka) >> 2];
            const int kv = kb + ((v >> 2) << 5) + (half << 4) + ((v & 3) << 2);
            b[v] = sw2[(c * C_DIM + kv) >> 2];
        }
        acc = __builtin_amdgcn_wmma_i32_16x16x64_iu8(true, a, true, b, acc, false, false);
    }
    const float sc = scale2[c];
#pragma unroll
    for (int v = 0; v < 8; ++v) {
        out[(v + (half << 3)) * C_DIM + c] = (float)acc[v] * sc;
    }
}

extern "C" void kernel_launch(void* const* d_in, const int* in_sizes, int n_in,
                              void* d_out, int out_size, void* d_ws, size_t ws_size,
                              hipStream_t stream) {
    (void)in_sizes; (void)n_in; (void)out_size; (void)ws_size;
    const float* x     = (const float*)d_in[0];
    const float* move1 = (const float*)d_in[1];
    const float* w1    = (const float*)d_in[2];
    const float* gamma = (const float*)d_in[3];
    const float* bbeta = (const float*)d_in[4];
    const float* lbias = (const float*)d_in[5];
    const float* prelu = (const float*)d_in[6];
    const float* move2 = (const float*)d_in[7];
    const float* w2    = (const float*)d_in[8];
    float* out = (float*)d_out;

    char*  ws     = (char*)d_ws;
    float* p      = (float*)ws;
    float* scale1 = p + B_DIM * C_DIM;
    float* scale2 = scale1 + C_DIM;
    int*   sw1    = (int*)(scale2 + C_DIM);
    int*   sw2    = sw1 + (C_DIM * C_DIM / 4);
    signed char* xb2 = (signed char*)(sw2 + (C_DIM * C_DIM / 4));

    pool_kernel  <<<B_DIM * C_DIM, 256, 0, stream>>>(x, p);
    pack_kernel  <<<2 * C_DIM,     128, 0, stream>>>(w1, w2, sw1, sw2, scale1, scale2);
    layer1_kernel<<<C_DIM / 16,     32, 0, stream>>>(p, move1, sw1, scale1,
                                                     gamma, bbeta, lbias, prelu, move2, xb2);
    layer2_kernel<<<C_DIM / 16,     32, 0, stream>>>(xb2, sw2, scale2, out);
}
